// MiniAttentionBlock_50964081935224
// MI455X (gfx1250) — matrix operand order repaired and checked by reading, not compiled, not run on hardware
//
#include <hip/hip_runtime.h>
#include <cstdint>

typedef __attribute__((ext_vector_type(16))) __bf16 v16bf;
typedef __attribute__((ext_vector_type(8)))  __bf16 v8bf;
typedef __attribute__((ext_vector_type(4)))  __bf16 v4bf;
typedef __attribute__((ext_vector_type(8)))  float  v8f;

#define FDIM    512
#define SDIM    4096
#define BATCH   16
#define M_TOTAL (BATCH * SDIM)   // 65536 rows
#define M_WG    64               // rows per workgroup (4 M-tiles of 16)
#define LDSTR   520              // padded LDS row stride in halves (1040 B -> bank spread)
#define EPSV    1e-7f

// ---------------------------------------------------------------------------
// Kernel 0: W [k][n] fp32 -> Wt [n][k] bf16 (contiguous-K B-fragments),
// zero exp-sum scalar and output (output is accumulated via atomics later).
// ---------------------------------------------------------------------------
__global__ __launch_bounds__(256) void prep_kernel(const float* __restrict__ W,
                                                   __bf16* __restrict__ Wt,
                                                   float* __restrict__ sum,
                                                   float* __restrict__ out) {
    int i = blockIdx.x * 256 + threadIdx.x;
    if (i < FDIM * FDIM) {
        int k = i >> 9;          // row of W
        int n = i & (FDIM - 1);  // col of W
        Wt[(size_t)n * FDIM + k] = (__bf16)W[i];
    }
    if (i < BATCH * FDIM) out[i] = 0.0f;
    if (i == 0) *sum = 0.0f;
}

// ---------------------------------------------------------------------------
// Kernel 1: fused  e[m] = exp( u . tanh(x[m,:] @ W + b) )  via bf16 WMMA,
// plus global atomic accumulation of sum(e).
//   - block = 256 threads = 8 waves, owns 64 rows (4 M-tiles)
//   - wave w owns N-slice [w*64, w*64+64) = 4 N-tiles
//   - x tile staged in LDS as bf16, B fragments loaded once per (n,k) and
//     reused across 4 M-tiles
// ---------------------------------------------------------------------------
__global__ __launch_bounds__(256) void scores_kernel(const float* __restrict__ x,
                                                     const __bf16* __restrict__ Wt,
                                                     const float* __restrict__ bias,
                                                     const float* __restrict__ u,
                                                     float* __restrict__ e_out,
                                                     float* __restrict__ sum) {
    extern __shared__ char smem[];
    __bf16* Abuf = (__bf16*)smem;                                   // 64 x 520 halves
    float*  sc   = (float*)(smem + M_WG * LDSTR * sizeof(__bf16));  // 64 row scores
    float*  red  = sc + M_WG;                                       // 8 wave partials

    const int tid  = threadIdx.x;
    const int lane = tid & 31;
    const int wv   = tid >> 5;
    const int m0   = blockIdx.x * M_WG;

    if (tid < M_WG) sc[tid] = 0.0f;

    // ---- stage x tile (64 rows x 512 f) fp32 -> bf16 LDS, coalesced float4 ----
    const float4* xs = (const float4*)(x + (size_t)m0 * FDIM);
#pragma unroll
    for (int it = 0; it < (M_WG * FDIM / 4) / 256; ++it) {
        int    idx = tid + it * 256;       // float4 index within tile
        float4 v   = xs[idx];
        int    row = idx >> 7;             // (idx*4)/512
        int    c4  = idx & 127;
        v4bf h;
        h[0] = (__bf16)v.x; h[1] = (__bf16)v.y;
        h[2] = (__bf16)v.z; h[3] = (__bf16)v.w;
        *(v4bf*)&Abuf[row * LDSTR + c4 * 4] = h;
    }
    __syncthreads();

    const int half = lane >> 4;   // 0: lanes 0-15, 1: lanes 16-31
    const int l15  = lane & 15;
    const int n0   = wv * 64;

    for (int nt = 0; nt < 4; ++nt) {
        const int ntb  = n0 + nt * 16;
        const int ncol = ntb + l15;        // this lane's N column

        v8f zero = {};
        v8f acc[4];
#pragma unroll
        for (int mt = 0; mt < 4; ++mt) acc[mt] = zero;

        // B fragment base: Wt row = ncol, 16 contiguous K-halves starting at half*16
        const __bf16* wp = Wt + (size_t)ncol * FDIM + half * 16;

        for (int k0 = 0; k0 < FDIM; k0 += 32) {
            v16bf bf = *(const v16bf*)(wp + k0);   // 32B aligned, 2x b128
#pragma unroll
            for (int mt = 0; mt < 4; ++mt) {
                // A fragment: lane holds row (mt*16+l15); two 8-half runs at
                // k0 + half*16  and  k0 + half*16 + 8  (contiguous K, matches B)
                const __bf16* ap = &Abuf[(mt * 16 + l15) * LDSTR + k0 + half * 16];
                v8bf lo = *(const v8bf*)ap;
                v8bf hi = *(const v8bf*)(ap + 8);
                v16bf af;
#pragma unroll
                for (int e = 0; e < 8; ++e) { af[e] = lo[e]; af[e + 8] = hi[e]; }
                acc[mt] = __builtin_amdgcn_wmma_f32_16x16x32_bf16(
                    false, af, false, bf, (short)0, acc[mt], false, false);
            }
        }

        // ---- epilogue: tanh(h + b) * u, reduce over N within each 16-lane half ----
        const float uval = u[ncol];
        const float bval = bias[ncol];
#pragma unroll
        for (int mt = 0; mt < 4; ++mt) {
#pragma unroll
            for (int r = 0; r < 8; ++r) {
                float hv = tanhf(acc[mt][r] + bval);
                float v  = hv * uval;
                v += __shfl_xor(v, 1, 32);
                v += __shfl_xor(v, 2, 32);
                v += __shfl_xor(v, 4, 32);
                v += __shfl_xor(v, 8, 32);
                // C layout: lanes 0-15 hold M=r, lanes 16-31 hold M=r+8
                if (l15 == 0) atomicAdd(&sc[mt * 16 + r + half * 8], v);
            }
        }
    }
    __syncthreads();

    // ---- exp + store e, block-reduce exp-sum, one global atomic per block ----
    float ev = 0.0f;
    if (tid < M_WG) {
        ev = expf(sc[tid]);
        e_out[m0 + tid] = ev;
    }
    ev += __shfl_xor(ev, 1, 32);
    ev += __shfl_xor(ev, 2, 32);
    ev += __shfl_xor(ev, 4, 32);
    ev += __shfl_xor(ev, 8, 32);
    ev += __shfl_xor(ev, 16, 32);
    if (lane == 0) red[wv] = ev;
    __syncthreads();
    if (tid == 0) {
        float t = 0.0f;
#pragma unroll
        for (int w = 0; w < 8; ++w) t += red[w];
        atomicAdd(sum, t);
    }
}

// ---------------------------------------------------------------------------
// Kernel 2: out[b,f] = (1/(sum+eps)) * sum_s x[b,s,f] * e[b,s]
// grid = (B, F/256, S/256); coalesced over f; partials via f32 global atomics.
// ---------------------------------------------------------------------------
__global__ __launch_bounds__(256) void pool_kernel(const float* __restrict__ x,
                                                   const float* __restrict__ e,
                                                   const float* __restrict__ sum,
                                                   float* __restrict__ out) {
    const int b  = blockIdx.x;
    const int f  = blockIdx.y * 256 + threadIdx.x;
    const int s0 = blockIdx.z * 256;

    const float* xb = x + ((size_t)b * SDIM + s0) * FDIM + f;
    const float* eb = e + b * SDIM + s0;

    float acc = 0.0f;
#pragma unroll 4
    for (int s = 0; s < 256; ++s)
        acc += xb[(size_t)s * FDIM] * eb[s];

    const float scale = 1.0f / (*sum + EPSV);
    atomicAdd(&out[b * FDIM + f], acc * scale);
}

// ---------------------------------------------------------------------------
extern "C" void kernel_launch(void* const* d_in, const int* in_sizes, int n_in,
                              void* d_out, int out_size, void* d_ws, size_t ws_size,
                              hipStream_t stream) {
    const float* x    = (const float*)d_in[0];   // [16,4096,512]
    const float* W    = (const float*)d_in[1];   // [512,512]
    const float* bias = (const float*)d_in[2];   // [512]
    const float* u    = (const float*)d_in[3];   // [512]
    float*       out  = (float*)d_out;           // [16,512]

    char*   ws  = (char*)d_ws;
    __bf16* Wt  = (__bf16*)ws;                                          // 512 KB
    float*  e   = (float*)(ws + (size_t)FDIM * FDIM * sizeof(__bf16));  // 256 KB
    float*  sum = (float*)(ws + (size_t)FDIM * FDIM * sizeof(__bf16)
                              + (size_t)M_TOTAL * sizeof(float));       // 4 B

    // 1) prepare Wt (bf16, transposed), zero sum + out
    prep_kernel<<<(FDIM * FDIM) / 256, 256, 0, stream>>>(W, Wt, sum, out);

    // 2) fused GEMM + tanh + u-dot + exp + global exp-sum
    const size_t smem = (size_t)M_WG * LDSTR * sizeof(__bf16)
                      + (M_WG + 8) * sizeof(float);                     // ~66.9 KB
    scores_kernel<<<M_TOTAL / M_WG, 256, smem, stream>>>(x, Wt, bias, u, e, sum);

    // 3) softmax-weighted pooling over S
    pool_kernel<<<dim3(BATCH, FDIM / 256, SDIM / 256), 256, 0, stream>>>(x, e, sum, out);
}
